// Glm4MoeNaiveMoeHybrid_1657857376742
// MI455X (gfx1250) — matrix operand order repaired and checked by reading, not compiled, not run on hardware
//
#include <hip/hip_runtime.h>
#include <hip/hip_bf16.h>

// Problem constants (match reference)
constexpr int E = 32;
constexpr int H = 2048;
constexpr int I = 1536;
constexpr int T = 1024;
constexpr int K = 8;
constexpr int TM = 64;          // token tile per block (4 WMMA M-subtiles)
constexpr int SUB = TM / 16;    // M-subtiles per wave
constexpr int CAP = T * K;      // worst-case (t,k) pairs one expert can receive
constexpr int MT = CAP / TM;    // max m-tiles per expert

typedef __attribute__((ext_vector_type(16))) __bf16 v16bf;
typedef __attribute__((ext_vector_type(8)))  float  v8f;
typedef __attribute__((ext_vector_type(4)))  float  f32x4;

union FragBF {
    v16bf v;
    uint4 q[2];
    unsigned u[8];
};

__device__ __forceinline__ unsigned pk2bf(float a, float b) {
    // truncate two f32 -> packed bf16x2 (lo, hi)
    unsigned ua = __float_as_uint(a);
    unsigned ub = __float_as_uint(b);
    return (ua >> 16) | (ub & 0xffff0000u);
}

__device__ __forceinline__ v8f vzero8() {
    v8f z;
#pragma unroll
    for (int i = 0; i < 8; ++i) z[i] = 0.0f;
    return z;
}

__device__ __forceinline__ v8f wmma_bf16(v16bf a, v16bf b, v8f c) {
    return __builtin_amdgcn_wmma_f32_16x16x32_bf16(
        /*neg_a=*/false, a, /*neg_b=*/false, b,
        /*c_mod=*/(short)0, c, /*reuse_a=*/false, /*reuse_b=*/false);
}

// A fragment (16x32 bf16) gathered from global fp32 rows (L2-resident x).
// rp already points at this lane's row + half-wave K skew (0 or 8 floats).
// Lanes 0-15: M=lane,   K chunks {k0..k0+7} in v0-3, {k0+16..k0+23} in v4-7
// Lanes16-31: M=lane-16, K chunks {k0+8..k0+15},     {k0+24..k0+31}
__device__ __forceinline__ v16bf load_a_x32(const float* rp, int k0) {
    f32x4 f0 = *(const f32x4*)(rp + k0 + 0);
    f32x4 f1 = *(const f32x4*)(rp + k0 + 4);
    f32x4 f2 = *(const f32x4*)(rp + k0 + 16);
    f32x4 f3 = *(const f32x4*)(rp + k0 + 20);
    FragBF f;
    f.u[0] = pk2bf(f0.x, f0.y);
    f.u[1] = pk2bf(f0.z, f0.w);
    f.u[2] = pk2bf(f1.x, f1.y);
    f.u[3] = pk2bf(f1.z, f1.w);
    f.u[4] = pk2bf(f2.x, f2.y);
    f.u[5] = pk2bf(f2.z, f2.w);
    f.u[6] = pk2bf(f3.x, f3.y);
    f.u[7] = pk2bf(f3.z, f3.w);
    return f.v;
}

// A fragment (16x32 bf16) from LDS (bf16, row-major, row_stride elements).
__device__ __forceinline__ v16bf load_a_lds(const __bf16* lds, int row_stride,
                                            int k0, int lane) {
    int m   = lane & 15;
    int klo = k0 + ((lane & 16) ? 8 : 0);
    FragBF f;
    f.q[0] = *(const uint4*)(lds + m * row_stride + klo);
    f.q[1] = *(const uint4*)(lds + m * row_stride + klo + 16);
    return f.v;
}

// B fragment (32x16 bf16) where logical B = W^T: B(k,n) = W[n0+n, k].
// Lane holds output column n = lane&15 (one weight row), K chunks:
// lanes 0-15 -> K={k0..k0+7, k0+16..k0+23}, lanes 16-31 -> K={k0+8..k0+15, k0+24..k0+31}.
// Weights are fp32, streamed once -> NON-TEMPORAL loads (keep L2 for x/h/out),
// truncate-packed to bf16 in-register.
__device__ __forceinline__ v16bf load_b_w32_nt(const float* mat, size_t row_stride,
                                               int n0, int k0, int lane) {
    int n = lane & 15;
    const float* r = mat + (size_t)(n0 + n) * row_stride + k0 + ((lane & 16) ? 8 : 0);
    __builtin_prefetch(r + 32, 0, 0);   // next K-chunk -> global_prefetch_b8
    f32x4 f0 = __builtin_nontemporal_load((const f32x4*)(r + 0));
    f32x4 f1 = __builtin_nontemporal_load((const f32x4*)(r + 4));
    f32x4 f2 = __builtin_nontemporal_load((const f32x4*)(r + 16));
    f32x4 f3 = __builtin_nontemporal_load((const f32x4*)(r + 20));
    FragBF f;
    f.u[0] = pk2bf(f0.x, f0.y);
    f.u[1] = pk2bf(f0.z, f0.w);
    f.u[2] = pk2bf(f1.x, f1.y);
    f.u[3] = pk2bf(f1.z, f1.w);
    f.u[4] = pk2bf(f2.x, f2.y);
    f.u[5] = pk2bf(f2.z, f2.w);
    f.u[6] = pk2bf(f3.x, f3.y);
    f.u[7] = pk2bf(f3.z, f3.w);
    return f.v;
}

// ---------------------------------------------------------------------------
// Kernel 0: zero the fp32 output (we accumulate into it with atomics).
__global__ void moe_zero_out(float* __restrict__ out) {
    int idx = blockIdx.x * blockDim.x + threadIdx.x;   // float4 index
    ((f32x4*)out)[idx] = (f32x4){0.f, 0.f, 0.f, 0.f};
}

// ---------------------------------------------------------------------------
// Kernel 1: deterministic per-expert compaction of routed (token, weight).
// One wave32 per expert scans all T*K pairs in fixed order.
__global__ void moe_compact(const int* __restrict__ topk_idx,
                            const float* __restrict__ topk_w,
                            int* __restrict__ counts,
                            int* __restrict__ tok_id,
                            float* __restrict__ tok_w) {
    const int e = blockIdx.x;
    const int lane = threadIdx.x;
    int cnt = 0;
    for (int base = 0; base < T * K; base += 32) {
        int p = base + lane;                 // T*K is a multiple of 32
        bool valid = (topk_idx[p] == e);
        unsigned long long mask = __ballot(valid);
        int pre = __popcll(mask & ((1ull << lane) - 1ull));
        if (valid) {
            tok_id[e * CAP + cnt + pre] = p / K;
            tok_w[e * CAP + cnt + pre]  = topk_w[p];
        }
        cnt += __popcll(mask);
    }
    if (lane == 0) counts[e] = cnt;
}

// ---------------------------------------------------------------------------
// Kernel 2: fused gate_up -> silu*up -> down for one (expert, 64-token tile).
// 256 threads = 8 waves. bf16 WMMA, fp32 accumulate, fp32 atomic output.
// LDS: h tile [64][1536] bf16 (192 KB) + token metadata.
__global__ __launch_bounds__(256)
void moe_expert(const float* __restrict__ x,
                const float* __restrict__ gup,
                const float* __restrict__ down,
                const int*  __restrict__ counts,
                const int*  __restrict__ tok_id,
                const float* __restrict__ tok_w,
                float* __restrict__ out) {
    extern __shared__ char smem[];
    __bf16* hs   = (__bf16*)smem;                      // [TM][I] bf16 : 192 KB
    int*    ttid = (int*)(smem + (size_t)TM * I * 2);  // [TM]
    float*  ttw  = (float*)(ttid + TM);                // [TM]

    const int e  = blockIdx.x / MT;
    const int mt = blockIdx.x % MT;
    const int cnt = counts[e];
    const int m0  = mt * TM;
    if (m0 >= cnt) return;                 // uniform across block: EXEC stays full

    const int tid  = threadIdx.x;
    const int lane = tid & 31;
    const int wave = tid >> 5;

    // Stage routed token ids & weights (pad tail slots with first token, w=0).
    if (tid < TM) {
        int mi = m0 + tid;
        bool v = (mi < cnt);
        int src = e * CAP + (v ? mi : m0);
        ttid[tid] = tok_id[src];
        ttw[tid]  = v ? tok_w[src] : 0.0f;
    }
    __syncthreads();

    // Per-lane x row pointers for the 4 M-subtiles (x stays hot in 192MB L2).
    const float* xrow[SUB];
#pragma unroll
    for (int s = 0; s < SUB; ++s) {
        int m = s * 16 + (lane & 15);
        xrow[s] = x + (size_t)ttid[m] * H + ((lane & 16) ? 8 : 0);
    }

    const float* gup_e  = gup  + (size_t)e * (2 * (size_t)I) * H;
    const float* up_e   = gup_e + (size_t)I * H;
    const float* down_e = down + (size_t)e * H * I;

    // ---- Phase 1: h = silu(x @ Wg^T) * (x @ Wu^T); wave owns 16 cols of I.
#pragma unroll 1
    for (int p = 0; p < I / 128; ++p) {          // 12 passes, 8 waves * 16 cols
        const int i0 = p * 128 + wave * 16;
        v8f accg[SUB], accu[SUB];
#pragma unroll
        for (int s = 0; s < SUB; ++s) { accg[s] = vzero8(); accu[s] = vzero8(); }
#pragma unroll 2
        for (int k0 = 0; k0 < H; k0 += 32) {
            v16bf bg = load_b_w32_nt(gup_e, H, i0, k0, lane);
            v16bf bu = load_b_w32_nt(up_e,  H, i0, k0, lane);
#pragma unroll
            for (int s = 0; s < SUB; ++s) {
                v16bf a = load_a_x32(xrow[s], k0);
                accg[s] = wmma_bf16(a, bg, accg[s]);
                accu[s] = wmma_bf16(a, bu, accu[s]);
            }
        }
        const int n   = lane & 15;
        const int mhi = (lane & 16) ? 8 : 0;
#pragma unroll
        for (int s = 0; s < SUB; ++s) {
#pragma unroll
            for (int r = 0; r < 8; ++r) {
                float g = accg[s][r];
                float u = accu[s][r];
                float hv = (g / (1.0f + __expf(-g))) * u;   // silu(g) * u
                hs[(s * 16 + r + mhi) * I + i0 + n] = (__bf16)hv;
            }
        }
    }
    __syncthreads();

    // ---- Phase 2: out[t, :] += w[t] * (h @ down_e^T); wave owns 16 out cols.
#pragma unroll 1
    for (int p = 0; p < H / 128; ++p) {          // 16 passes
        const int hc = p * 128 + wave * 16;
        v8f acc[SUB];
#pragma unroll
        for (int s = 0; s < SUB; ++s) acc[s] = vzero8();
#pragma unroll 2
        for (int k0 = 0; k0 < I; k0 += 32) {
            v16bf b = load_b_w32_nt(down_e, I, hc, k0, lane);
#pragma unroll
            for (int s = 0; s < SUB; ++s) {
                v16bf a = load_a_lds(hs + (size_t)s * 16 * I, I, k0, lane);
                acc[s] = wmma_bf16(a, b, acc[s]);
            }
        }
        const int n   = lane & 15;
        const int mhi = (lane & 16) ? 8 : 0;
#pragma unroll
        for (int s = 0; s < SUB; ++s) {
#pragma unroll
            for (int r = 0; r < 8; ++r) {
                int m = s * 16 + r + mhi;
                float val = acc[s][r] * ttw[m];
                atomicAdd(out + (size_t)ttid[m] * H + hc + n, val);
            }
        }
    }
}

// ---------------------------------------------------------------------------
extern "C" void kernel_launch(void* const* d_in, const int* in_sizes, int n_in,
                              void* d_out, int out_size, void* d_ws, size_t ws_size,
                              hipStream_t stream) {
    const float* hidden = (const float*)d_in[0];   // (T, H)
    const int*   tki    = (const int*)d_in[1];     // (T, K)
    const float* tkw    = (const float*)d_in[2];   // (T, K)
    const float* gup    = (const float*)d_in[3];   // (E, 2I, H)
    const float* down   = (const float*)d_in[4];   // (E, H, I)
    float* out = (float*)d_out;                    // (T, H)

    // Workspace layout
    char* ws = (char*)d_ws;
    int*   counts = (int*)ws;                                 // E ints
    int*   tokid  = (int*)(ws + 256);                         // E*CAP ints (1 MB)
    float* tokw   = (float*)(ws + 256 + (size_t)E * CAP * 4); // E*CAP floats (1 MB)

    // 1) zero output (accumulated via atomics)
    moe_zero_out<<<(T * H / 4) / 256, 256, 0, stream>>>(out);

    // 2) deterministic per-expert routing compaction
    moe_compact<<<E, 32, 0, stream>>>(tki, tkw, counts, tokid, tokw);

    // 3) fused expert MLP over (expert, 64-token-tile) grid
    const size_t smem_bytes = (size_t)TM * I * 2
                            + TM * sizeof(int) + TM * sizeof(float);
    moe_expert<<<E * MT, 256, smem_bytes, stream>>>(hidden, gup, down,
                                                    counts, tokid, tokw, out);
}
